// CrossAttention_38448547234773
// MI455X (gfx1250) — matrix operand order repaired and checked by reading, not compiled, not run on hardware
//
#include <hip/hip_runtime.h>

// ---------------------------------------------------------------------------
// Cross-attention forward for MI455X (gfx1250, wave32, WMMA + TDM).
// cvt(f32->bf16) -> TDM-pipelined WMMA GEMM (Q) -> RMSNorm(q) ->
// TDM-pipelined WMMA GEMM (KV) -> RMSNorm(k)+split(v^T) ->
// TDM-staged flash-attention with v_wmma_f32_16x16x32_bf16.
// All LDS tiles use TDM pad (4 dwords per 32) -> row stride 72 elems, making
// the 16-lane ds_load_b128 fragment reads bank-conflict-free.
// ---------------------------------------------------------------------------

#define DIMX 1024
#define HEADS 16
#define DH 64
#define BATCH 2
#define SEQN 2048
#define SEQM 2048
#define EPS 1.1920928955078125e-07f
#define LSTR 72   // padded LDS row stride (elems) for 64-elem tile rows

typedef __attribute__((ext_vector_type(16))) __bf16 v16bf;
typedef __attribute__((ext_vector_type(8)))  float v8f;
typedef __attribute__((ext_vector_type(4)))  unsigned int v4u;
typedef __attribute__((ext_vector_type(8)))  int v8i;
typedef __attribute__((ext_vector_type(4)))  int v4i;

#if __has_builtin(__builtin_amdgcn_tensor_load_to_lds)
#define HAVE_TDM 1
#else
#define HAVE_TDM 0
#endif

static __device__ __forceinline__ unsigned short f32_to_bf16(float f) {
  unsigned int u = __float_as_uint(f);
  u += 0x7FFFu + ((u >> 16) & 1u);   // round-to-nearest-even
  return (unsigned short)(u >> 16);
}

static __device__ __forceinline__ v16bf make_frag(uint4 lo, uint4 hi) {
  union { uint4 u[2]; v16bf v; } c;
  c.u[0] = lo; c.u[1] = hi;
  return c.v;
}

// A-matrix 16x32 bf16 fragment (MxK), row-major source, `row_stride` elems:
// lane<16:  row M=lane,    K = {k0..k0+7, k0+16..k0+23}
// lane>=16: row M=lane-16, K = {k0+8..k0+15, k0+24..k0+31}
static __device__ __forceinline__ v16bf load_a_frag(const unsigned short* base,
                                                    int row_stride, int k0) {
  const int lane = threadIdx.x & 31;
  const int hf = lane >> 4;
  const int r  = lane & 15;
  const unsigned short* p = base + (long)r * row_stride + k0 + hf * 8;
  uint4 lo = *(const uint4*)(p);
  uint4 hi = *(const uint4*)(p + 16);
  return make_frag(lo, hi);
}

// B-matrix 32x16 bf16 fragment (KxN), element (k,n) = rowmaj[n][k] (K contig):
// lane<16:  col N=lane,    K = {k0..k0+7, k0+16..k0+23}
// lane>=16: col N=lane-16, K = {k0+8..k0+15, k0+24..k0+31}
static __device__ __forceinline__ v16bf load_b_frag(const unsigned short* base,
                                                    int row_stride, int k0) {
  const int lane = threadIdx.x & 31;
  const int hf = lane >> 4;
  const int n  = lane & 15;
  const unsigned short* p = base + (long)n * row_stride + k0 + hf * 8;
  uint4 lo = *(const uint4*)(p);
  uint4 hi = *(const uint4*)(p + 16);
  return make_frag(lo, hi);
}

static __device__ __forceinline__ v8f wmma_bf16(v16bf a, v16bf b, v8f c) {
  return __builtin_amdgcn_wmma_f32_16x16x32_bf16(false, a, false, b,
                                                 (short)0, c, false, false);
}

static __device__ __forceinline__ unsigned lds_off(const void* p) {
  // generic pointer to __shared__: low 32 bits hold the LDS byte offset
  return (unsigned)(unsigned long long)(uintptr_t)p;
}

#if HAVE_TDM
// TDM 2D tile load (bf16): 64 x `rows` elements, global row stride
// `stride0` elems, into LDS at `ldsa` with 4-dword pad per 32 dwords
// (=> LDS row stride LSTR=72 elems).  D# per CDNA5 ISA.
static __device__ __forceinline__ void tdm_load_tile64(unsigned ldsa, const void* gaddr,
                                                       unsigned rows, unsigned stride0) {
  unsigned long long ga = (unsigned long long)(uintptr_t)gaddr;
  v4u g0;
  g0[0] = 1u;                                            // count=1 (valid D#)
  g0[1] = ldsa;                                          // lds_addr
  g0[2] = (unsigned)(ga & 0xFFFFFFFFu);                  // global_addr[31:0]
  g0[3] = (unsigned)((ga >> 32) & 0x01FFFFFFu) | (2u << 30);  // [56:32] | type=2
  v8i g1;
  // data_size=1(2B) | pad_enable | pad_interval=4(32 dw) | pad_amount=3(4 dw)
  g1[0] = (int)((1u << 16) | (1u << 20) | (4u << 22) | (3u << 25));
  g1[1] = (int)((64u & 0xFFFFu) << 16);                  // tensor_dim0 = 64
  g1[2] = (int)((rows & 0xFFFFu) << 16);                 // tensor_dim1 = rows
  g1[3] = (int)(((rows >> 16) & 0xFFFFu) | (64u << 16)); // | tile_dim0 = 64
  g1[4] = (int)(rows & 0xFFFFu);                         // tile_dim1 (tile_dim2=0)
  g1[5] = (int)stride0;                                  // tensor_dim0_stride
  g1[6] = 0;
  g1[7] = 0;
  v4i z4 = {0, 0, 0, 0};
#if defined(__clang_major__) && (__clang_major__ >= 23)
  v8i z8 = {0, 0, 0, 0, 0, 0, 0, 0};
  __builtin_amdgcn_tensor_load_to_lds(g0, g1, z4, z4, z8, 0);
#else
  __builtin_amdgcn_tensor_load_to_lds(g0, g1, z4, z4, 0);
#endif
}
#endif

// ---------------------------------------------------------------------------
__global__ void cvt_f32_bf16(const float* __restrict__ src,
                             unsigned short* __restrict__ dst, int count) {
  int i = blockIdx.x * blockDim.x + threadIdx.x;
  int stride = gridDim.x * blockDim.x;
  for (; i < count; i += stride) dst[i] = f32_to_bf16(src[i]);
}

// C[M x N] = A[M x K] @ W[N x K]^T + bias.  A,W bf16 row-major (K contiguous).
// 256 threads (8 waves) -> 128 rows x 64 cols.  K in 64-wide steps staged in
// double-buffered padded LDS tiles by the TDM; fragments are batch-loaded so
// the 8 WMMAs of a step issue back-to-back behind one s_wait_dscnt.
__global__ void gemm_bias_wmma(const unsigned short* __restrict__ A,
                               const unsigned short* __restrict__ W,
                               const float* __restrict__ bias,
                               float* __restrict__ C, int M, int N, int K) {
  __shared__ unsigned short As[2][128 * LSTR];   // 36 KB
  __shared__ unsigned short Ws[2][64 * LSTR];    // 18 KB
  const int lane = threadIdx.x & 31;
  const int wave = threadIdx.x >> 5;
  const int row0 = blockIdx.y * 128 + wave * 16;
  const int n0   = blockIdx.x * 64;
  const unsigned short* abase = A + ((long)blockIdx.y * 128) * K;
  const unsigned short* wbase = W + (long)n0 * K;
  const int steps = K >> 6;

  v8f acc[4] = {};

#if HAVE_TDM
  if (wave == 0) {
    tdm_load_tile64(lds_off(&As[0][0]), abase, 128, K);
    tdm_load_tile64(lds_off(&Ws[0][0]), wbase, 64,  K);
    __builtin_amdgcn_s_wait_tensorcnt(0);
  }
  __syncthreads();
  if (wave == 0 && steps > 1) {
    tdm_load_tile64(lds_off(&As[1][0]), abase + 64, 128, K);
    tdm_load_tile64(lds_off(&Ws[1][0]), wbase + 64, 64,  K);
  }
#endif
  for (int kt = 0; kt < steps; ++kt) {
#if HAVE_TDM
    const int buf = kt & 1;
#else
    const int buf = 0;
    {   // cooperative staging with the same padded layout
      const int k0 = kt * 64;
      for (int v = threadIdx.x; v < 1024; v += 256) {     // A: 128 rows x 8 vec
        const int r = v >> 3, c8 = (v & 7) * 8;
        *(uint4*)&As[0][r * LSTR + c8] = *(const uint4*)(abase + (long)r * K + k0 + c8);
      }
      for (int v = threadIdx.x; v < 512; v += 256) {      // W: 64 rows x 8 vec
        const int r = v >> 3, c8 = (v & 7) * 8;
        *(uint4*)&Ws[0][r * LSTR + c8] = *(const uint4*)(wbase + (long)r * K + k0 + c8);
      }
      __syncthreads();
    }
#endif
    // batch all fragment loads, then 8 back-to-back WMMAs
    v16bf a0 = load_a_frag(&As[buf][wave * 16 * LSTR], LSTR, 0);
    v16bf a1 = load_a_frag(&As[buf][wave * 16 * LSTR], LSTR, 32);
    v16bf w[8];
#pragma unroll
    for (int ct = 0; ct < 4; ++ct) {
      w[ct]     = load_b_frag(&Ws[buf][ct * 16 * LSTR], LSTR, 0);
      w[4 + ct] = load_b_frag(&Ws[buf][ct * 16 * LSTR], LSTR, 32);
    }
#pragma unroll
    for (int ct = 0; ct < 4; ++ct) acc[ct] = wmma_bf16(a0, w[ct], acc[ct]);
#pragma unroll
    for (int ct = 0; ct < 4; ++ct) acc[ct] = wmma_bf16(a1, w[4 + ct], acc[ct]);

#if HAVE_TDM
    if (wave == 0) __builtin_amdgcn_s_wait_tensorcnt(0);  // kt+1 tiles landed
    __syncthreads();                                      // readers done w/ buf
    if (wave == 0 && kt + 2 < steps) {
      const int k2 = (kt + 2) * 64;
      tdm_load_tile64(lds_off(&As[buf][0]), abase + k2, 128, K);
      tdm_load_tile64(lds_off(&Ws[buf][0]), wbase + k2, 64,  K);
    }
#else
    __syncthreads();
#endif
  }

  const int hf = lane >> 4, cl = lane & 15;
#pragma unroll
  for (int ct = 0; ct < 4; ++ct) {
    const int col = n0 + ct * 16 + cl;
    const float bv = bias[col];
#pragma unroll
    for (int r = 0; r < 8; ++r) {
      const int row = row0 + r + 8 * hf;
      C[(long)row * N + col] = acc[ct][r] + bv;
    }
  }
}

// Per-(row,head) RMSNorm of q, write head-major bf16 [bh][n][d].
__global__ void qnorm_kernel(const float* __restrict__ qf,
                             const float* __restrict__ gamma,
                             unsigned short* __restrict__ qb) {
  const int gid  = blockIdx.x * (blockDim.x >> 5) + (threadIdx.x >> 5);
  const int lane = threadIdx.x & 31;
  const int h  = gid & (HEADS - 1);
  const int rn = gid >> 4;                       // b*SEQN + n
  const float* src = qf + (long)rn * DIMX + h * DH;
  const float v0 = src[lane], v1 = src[lane + 32];
  float ss = v0 * v0 + v1 * v1;
#pragma unroll
  for (int off = 16; off; off >>= 1) ss += __shfl_xor(ss, off, 32);
  const float r = rsqrtf(ss * (1.0f / DH) + EPS);
  const int b = rn >> 11, n = rn & (SEQN - 1);
  unsigned short* dst = qb + ((long)(b * HEADS + h) * SEQN + n) * DH;
  dst[lane]      = f32_to_bf16(v0 * r * gamma[lane]);
  dst[lane + 32] = f32_to_bf16(v1 * r * gamma[lane + 32]);
}

// Split interleaved kv, RMSNorm k -> [bh][m][d]; v transposed -> [bh][d][m].
__global__ void kvnorm_kernel(const float* __restrict__ kvf,
                              const float* __restrict__ kgamma,
                              unsigned short* __restrict__ kb,
                              unsigned short* __restrict__ vb) {
  const int gid  = blockIdx.x * (blockDim.x >> 5) + (threadIdx.x >> 5);
  const int lane = threadIdx.x & 31;
  const int h  = gid & (HEADS - 1);
  const int rn = gid >> 4;                       // b*SEQM + m
  const float* src = kvf + (long)rn * (2 * DIMX) + h * (2 * DH);
  const float k0 = src[2 * lane],       v0 = src[2 * lane + 1];
  const float k1 = src[2 * lane + 64],  v1 = src[2 * lane + 65];
  float ss = k0 * k0 + k1 * k1;
#pragma unroll
  for (int off = 16; off; off >>= 1) ss += __shfl_xor(ss, off, 32);
  const float r = rsqrtf(ss * (1.0f / DH) + EPS);
  const int b = rn >> 11, m = rn & (SEQM - 1);
  const long bh = b * HEADS + h;
  unsigned short* kd = kb + (bh * SEQM + m) * DH;
  kd[lane]      = f32_to_bf16(k0 * r * kgamma[lane]);
  kd[lane + 32] = f32_to_bf16(k1 * r * kgamma[lane + 32]);
  unsigned short* vd = vb + bh * DH * SEQM;
  vd[(long)lane * SEQM + m]        = f32_to_bf16(v0);
  vd[(long)(lane + 32) * SEQM + m] = f32_to_bf16(v1);
}

// Flash attention: block = 4 waves, 64 q-rows, one (b,h).  K/V staged per
// 64-wide m-chunk into double-buffered padded LDS by the TDM (shared by all
// 4 waves instead of 4x redundant global loads).  Per chunk: 8 score WMMAs,
// online softmax, P through LDS, 8 AV WMMAs; next chunk's DMA overlaps.
__global__ void attn_kernel(const unsigned short* __restrict__ qb,
                            const unsigned short* __restrict__ kb,
                            const unsigned short* __restrict__ vb,
                            float* __restrict__ out) {
  __shared__ unsigned short Ks[2][64 * LSTR];    // 18 KB
  __shared__ unsigned short Vs[2][64 * LSTR];    // 18 KB
  __shared__ unsigned short Ps[4][16 * LSTR];    // 9 KB (per-wave P tiles)
  const int lane = threadIdx.x & 31;
  const int wave = threadIdx.x >> 5;
  const int nt = blockIdx.x & 31;                // SEQN/64 tiles
  const int bh = blockIdx.x >> 5;
  const int b = bh >> 4, h = bh & (HEADS - 1);
  const int row0 = nt * 64 + wave * 16;

  const unsigned short* qbase = qb + ((long)bh * SEQN + row0) * DH;
  const unsigned short* kbase = kb + (long)bh * SEQM * DH;
  const unsigned short* vbase = vb + (long)bh * DH * SEQM;

  const v16bf a0 = load_a_frag(qbase, DH, 0);
  const v16bf a1 = load_a_frag(qbase, DH, 32);

  v8f acc[4] = {};
  float m_i[8], l_i[8];
#pragma unroll
  for (int r = 0; r < 8; ++r) { m_i[r] = -1e30f; l_i[r] = 0.0f; }

  const int hf = lane >> 4, cl = lane & 15;
  const float scale = 0.125f;                    // 1/sqrt(64)
  const int steps = SEQM / 64;                   // 32

#if HAVE_TDM
  if (wave == 0) {
    tdm_load_tile64(lds_off(&Ks[0][0]), kbase, 64, DH);    // K rows m0..m0+63
    tdm_load_tile64(lds_off(&Vs[0][0]), vbase, 64, SEQM);  // V cols m0..m0+63
    __builtin_amdgcn_s_wait_tensorcnt(0);
  }
  __syncthreads();
  if (wave == 0) {
    tdm_load_tile64(lds_off(&Ks[1][0]), kbase + (long)64 * DH, 64, DH);
    tdm_load_tile64(lds_off(&Vs[1][0]), vbase + 64, 64, SEQM);
  }
#endif
  for (int mt = 0; mt < steps; ++mt) {
#if HAVE_TDM
    const int buf = mt & 1;
#else
    const int buf = 0;
    {   // cooperative staging with the same padded layout
      const int m0 = mt * 64;
      for (int v = threadIdx.x; v < 512; v += 128) {
        const int r = v >> 3, c8 = (v & 7) * 8;
        *(uint4*)&Ks[0][r * LSTR + c8] = *(const uint4*)(kbase + (long)(m0 + r) * DH + c8);
        *(uint4*)&Vs[0][r * LSTR + c8] = *(const uint4*)(vbase + (long)r * SEQM + m0 + c8);
      }
      __syncthreads();
    }
#endif
    // ---- S = (Q K^T): batch 8 B-frags then 8 WMMAs -----------------------
    v16bf kf[8];
#pragma unroll
    for (int t = 0; t < 4; ++t) {
      kf[2 * t]     = load_b_frag(&Ks[buf][t * 16 * LSTR], LSTR, 0);
      kf[2 * t + 1] = load_b_frag(&Ks[buf][t * 16 * LSTR], LSTR, 32);
    }
    v8f s[4];
#pragma unroll
    for (int t = 0; t < 4; ++t) {
      v8f cs = {};
      cs = wmma_bf16(a0, kf[2 * t], cs);
      cs = wmma_bf16(a1, kf[2 * t + 1], cs);
      s[t] = cs;
    }

    // ---- online softmax (row stats reduced over 16-lane halves) ----------
#pragma unroll
    for (int r = 0; r < 8; ++r) {
      float sv[4];
#pragma unroll
      for (int t = 0; t < 4; ++t) sv[t] = s[t][r] * scale;
      float mx = fmaxf(fmaxf(sv[0], sv[1]), fmaxf(sv[2], sv[3]));
#pragma unroll
      for (int off = 8; off; off >>= 1) mx = fmaxf(mx, __shfl_xor(mx, off, 16));
      const float mn = fmaxf(m_i[r], mx);
      const float corr = __expf(m_i[r] - mn);
      m_i[r] = mn;
      float rs = 0.0f;
      const int prow = (r + 8 * hf) * LSTR;
#pragma unroll
      for (int t = 0; t < 4; ++t) {
        const float p = __expf(sv[t] - mn);
        rs += p;
        Ps[wave][prow + t * 16 + cl] = f32_to_bf16(p);
      }
#pragma unroll
      for (int off = 8; off; off >>= 1) rs += __shfl_xor(rs, off, 16);
      l_i[r] = l_i[r] * corr + rs;
#pragma unroll
      for (int ct = 0; ct < 4; ++ct) acc[ct][r] *= corr;
    }
    asm volatile("s_wait_dscnt 0" ::: "memory");

    // ---- O += P V: batch P/V frags then 8 WMMAs --------------------------
    v16bf pa0 = load_a_frag(&Ps[wave][0], LSTR, 0);
    v16bf pa1 = load_a_frag(&Ps[wave][0], LSTR, 32);
    v16bf vf[8];
#pragma unroll
    for (int ct = 0; ct < 4; ++ct) {
      vf[2 * ct]     = load_b_frag(&Vs[buf][ct * 16 * LSTR], LSTR, 0);
      vf[2 * ct + 1] = load_b_frag(&Vs[buf][ct * 16 * LSTR], LSTR, 32);
    }
#pragma unroll
    for (int ct = 0; ct < 4; ++ct) acc[ct] = wmma_bf16(pa0, vf[2 * ct], acc[ct]);
#pragma unroll
    for (int ct = 0; ct < 4; ++ct) acc[ct] = wmma_bf16(pa1, vf[2 * ct + 1], acc[ct]);

#if HAVE_TDM
    if (wave == 0) __builtin_amdgcn_s_wait_tensorcnt(0);   // mt+1 chunk landed
    __syncthreads();                                       // Ps + buf reads done
    if (wave == 0 && mt + 2 < steps) {
      const long m2 = (long)(mt + 2) * 64;
      tdm_load_tile64(lds_off(&Ks[buf][0]), kbase + m2 * DH, 64, DH);
      tdm_load_tile64(lds_off(&Vs[buf][0]), vbase + m2, 64, SEQM);
    }
#else
    __syncthreads();
#endif
  }

  // ---- epilogue: normalize by l_i, scatter to [b][n][h*64+d] -------------
#pragma unroll
  for (int r = 0; r < 8; ++r) {
    const float inv = 1.0f / l_i[r];
    const int n = row0 + r + 8 * hf;
#pragma unroll
    for (int ct = 0; ct < 4; ++ct) {
      const int d = ct * 16 + cl;
      out[(long)b * SEQN * DIMX + (long)n * DIMX + h * DH + d] = acc[ct][r] * inv;
    }
  }
}

// ---------------------------------------------------------------------------
extern "C" void kernel_launch(void* const* d_in, const int* in_sizes, int n_in,
                              void* d_out, int out_size, void* d_ws, size_t ws_size,
                              hipStream_t stream) {
  (void)in_sizes; (void)n_in; (void)out_size; (void)ws_size;
  const float* x   = (const float*)d_in[0];
  const float* c   = (const float*)d_in[1];
  const float* Wq  = (const float*)d_in[2];
  const float* bq  = (const float*)d_in[3];
  const float* Wkv = (const float*)d_in[4];
  const float* bkv = (const float*)d_in[5];
  const float* qg  = (const float*)d_in[6];
  const float* kg  = (const float*)d_in[7];
  float* out = (float*)d_out;

  const long XBE  = (long)BATCH * SEQN * DIMX;
  const long CBE  = (long)BATCH * SEQM * DIMX;
  const long WQE  = (long)DIMX * DIMX;
  const long WKVE = (long)2 * DIMX * DIMX;
  const long HDE  = (long)BATCH * HEADS * SEQN * DH;

  unsigned short* xb   = (unsigned short*)d_ws;
  unsigned short* cb   = xb + XBE;
  unsigned short* wqb  = cb + CBE;
  unsigned short* wkvb = wqb + WQE;
  unsigned short* qbuf = wkvb + WKVE;
  unsigned short* kbuf = qbuf + HDE;
  unsigned short* vbuf = kbuf + HDE;
  float* scr = (float*)(vbuf + HDE);                 // reused: qf32 then kvf32

  cvt_f32_bf16<<<4096, 256, 0, stream>>>(x,   xb,   (int)XBE);
  cvt_f32_bf16<<<4096, 256, 0, stream>>>(c,   cb,   (int)CBE);
  cvt_f32_bf16<<<1024, 256, 0, stream>>>(Wq,  wqb,  (int)WQE);
  cvt_f32_bf16<<<2048, 256, 0, stream>>>(Wkv, wkvb, (int)WKVE);

  const int Mrows = BATCH * SEQN;                    // 4096
  gemm_bias_wmma<<<dim3(DIMX / 64, Mrows / 128), 256, 0, stream>>>(
      xb, wqb, bq, scr, Mrows, DIMX, DIMX);
  qnorm_kernel<<<(Mrows * HEADS) / 8, 256, 0, stream>>>(scr, qg, qbuf);

  gemm_bias_wmma<<<dim3((2 * DIMX) / 64, Mrows / 128), 256, 0, stream>>>(
      cb, wkvb, bkv, scr, Mrows, 2 * DIMX, DIMX);
  kvnorm_kernel<<<(Mrows * HEADS) / 8, 256, 0, stream>>>(scr, kg, kbuf, vbuf);

  attn_kernel<<<BATCH * HEADS * (SEQN / 64), 128, 0, stream>>>(
      qbuf, kbuf, vbuf, out);
}
